// PredictorPerCT_21457656611114
// MI455X (gfx1250) — matrix operand order repaired and checked by reading, not compiled, not run on hardware
//
#include <hip/hip_runtime.h>
#include <hip/hip_bf16.h>

// Problem constants (match reference).
#define N_SAMPLES 16384
#define DIM_D     1024
#define DIM_H     4096
#define NUM_E     8
#define TMROW     64            // rows per GEMM block tile (4 x WMMA M)
#define TN        256           // cols per block (8 col-groups x 32)
#define TK        32            // WMMA K per step (bf16)
#define NTHREADS  512           // 16 wave32: (m-half, col-group)
#define KCHUNK    1024          // A-panel K chunk staged in LDS
#define AKP       (KCHUNK + 8)  // LDS A row stride (bf16 elems), padded
#define NPAD      (N_SAMPLES + NUM_E * TMROW)  // 16896 worst-case padded rows
#define MAX_TILES (NPAD / TMROW)               // 264

typedef __bf16 bf16_t;
typedef __attribute__((ext_vector_type(16))) __bf16 v16bf;
typedef __attribute__((ext_vector_type(8)))  __bf16 v8bf;
typedef __attribute__((ext_vector_type(4)))  __bf16 v4bf;
typedef __attribute__((ext_vector_type(8)))  float  v8f;

// ---------------- precision / layout preprocessing ----------------

// x (f32 [N,D]) -> xb (bf16 [N,D]); vectorized 4-wide.
__global__ void convert_x_kernel(const float4* __restrict__ x4,
                                 v4bf* __restrict__ xb4, int n4) {
    int i = blockIdx.x * blockDim.x + threadIdx.x;
    if (i >= n4) return;
    float4 v = x4[i];
    v4bf o;
    o[0] = (bf16_t)v.x; o[1] = (bf16_t)v.y;
    o[2] = (bf16_t)v.z; o[3] = (bf16_t)v.w;
    xb4[i] = o;
}

// W (f32 [E,K,N]) -> WT (bf16 [E,N,K]) via 32x32 LDS tile transpose.
__global__ void convert_transpose_kernel(const float* __restrict__ W,
                                         bf16_t* __restrict__ WT,
                                         int K, int Ndim) {
    __shared__ float tile[32][33];
    const int e  = blockIdx.z;
    const int k0 = blockIdx.x * 32;
    const int n0 = blockIdx.y * 32;
    const int tx = threadIdx.x;      // 0..31
    const int ty = threadIdx.y;      // 0..7
    const size_t base = (size_t)e * K * Ndim;
#pragma unroll
    for (int r = 0; r < 4; ++r) {
        int k = ty * 4 + r;
        tile[k][tx] = W[base + (size_t)(k0 + k) * Ndim + n0 + tx];
    }
    __syncthreads();
#pragma unroll
    for (int r = 0; r < 4; ++r) {
        int n = ty * 4 + r;
        WT[base + (size_t)(n0 + n) * K + k0 + tx] = (bf16_t)tile[tx][n];
    }
}

// ---------------- routing / dispatch kernels ----------------

__global__ void init_counts_kernel(int* __restrict__ counts) {
    if (threadIdx.x < NUM_E) counts[threadIdx.x] = 0;
}

__global__ void route_kernel(const float* __restrict__ tct,
                             int* __restrict__ route,
                             int* __restrict__ counts) {
    int n = blockIdx.x * blockDim.x + threadIdx.x;
    if (n >= N_SAMPLES) return;
    const float* r = tct + (size_t)n * NUM_E;
    float best = r[0];
    int bi = 0;
#pragma unroll
    for (int j = 1; j < NUM_E; ++j) {
        float v = r[j];
        if (v > best) { best = v; bi = j; }   // strict > keeps first max (jnp.argmax)
    }
    route[n] = bi;
    atomicAdd(&counts[bi], 1);
}

__global__ void perm_init_kernel(int* __restrict__ perm) {
    int i = blockIdx.x * blockDim.x + threadIdx.x;
    if (i < NPAD) perm[i] = -1;
}

__global__ void scan_kernel(const int* __restrict__ counts,
                            int* __restrict__ poff,
                            int* __restrict__ cursors) {
    if (threadIdx.x != 0 || blockIdx.x != 0) return;
    int acc = 0;
#pragma unroll
    for (int e = 0; e < NUM_E; ++e) {
        poff[e] = acc;
        cursors[e] = acc;
        int c = counts[e];
        acc += ((c + TMROW - 1) / TMROW) * TMROW;  // pad segment to tile multiple
    }
    poff[NUM_E] = acc;
}

__global__ void scatter_kernel(const int* __restrict__ route,
                               int* __restrict__ cursors,
                               int* __restrict__ perm) {
    int n = blockIdx.x * blockDim.x + threadIdx.x;
    if (n >= N_SAMPLES) return;
    int pos = atomicAdd(&cursors[route[n]], 1);
    perm[pos] = n;
}

// ---------------- grouped GEMM (WMMA bf16, f32 accumulate) ----------------
// Block: 512 threads = 16 wave32. Tile: 64 (M) x 256 (N), K-step 32.
// Wave w: m-half = w>>3 (rows 0-31 / 32-63), col-group = w&7 (32 cols).
// A panel (64 x KCHUNK bf16, gathered rows) staged to LDS with
// global_load_async_to_lds_b128 (ASYNCcnt). B fragments read directly from
// L2-resident transposed bf16 weights (32B contiguous per lane). 4 WMMAs per
// wave per K-step, zero inner-loop barriers.

template <int K_TOTAL, bool RELU, bool OUT_BF16>
__launch_bounds__(NTHREADS)
__global__ void moe_gemm_kernel(const bf16_t* __restrict__ A,     // [*, K_TOTAL]
                                const bf16_t* __restrict__ WT,    // [E, NDIM, K_TOTAL]
                                const float*  __restrict__ Bias,  // [E, NDIM]
                                const int*    __restrict__ perm,
                                const int*    __restrict__ poff,
                                bf16_t*       __restrict__ outB,
                                float*        __restrict__ outF,
                                int NDIM) {
    extern __shared__ char smem[];
    bf16_t* As = (bf16_t*)smem;     // TMROW x AKP (~129 KB)
    __shared__ int rows_s[TMROW];

    const int tid = threadIdx.x;
    const int rowbase = blockIdx.x * TMROW;
    if (rowbase >= poff[NUM_E]) return;       // uniform per-block exit

    int e = 0;
#pragma unroll
    for (int i = 0; i < NUM_E - 1; ++i)
        if (rowbase >= poff[i + 1]) e = i + 1;

    if (tid < TMROW) rows_s[tid] = perm[rowbase + tid];
    __syncthreads();

    const int colbase = blockIdx.y * TN;
    const int lane  = tid & 31;
    const int wv    = tid >> 5;       // 0..15
    const int cg    = wv & 7;         // col group
    const int mbase = (wv >> 3) * 32; // 0 or 32
    const int l15   = lane & 15;
    const int khalf = lane >> 4;
    const int c0 = cg * 32 + l15;
    const int c1 = c0 + 16;

    // Per-lane B column pointers (transposed weights: contiguous in K).
    const size_t Wexp = (size_t)e * NDIM * K_TOTAL;
    const bf16_t* pb0 = WT + Wexp + (size_t)(colbase + c0) * K_TOTAL + khalf * 16;
    const bf16_t* pb1 = WT + Wexp + (size_t)(colbase + c1) * K_TOTAL + khalf * 16;

    const float bias0 = Bias[(size_t)e * NDIM + colbase + c0];
    const float bias1 = Bias[(size_t)e * NDIM + colbase + c1];

    v8f acc00 = {}, acc01 = {}, acc10 = {}, acc11 = {};

#pragma unroll 1
    for (int kc = 0; kc < K_TOTAL; kc += KCHUNK) {
        // ---- Stage A panel: TMROW x KCHUNK bf16, 16B per lane per op ----
        for (int i = tid; i < TMROW * (KCHUNK / 8); i += NTHREADS) {
            int m  = i >> 7;           // KCHUNK/8 == 128 chunks per row
            int ch = i & 127;
            int row = rows_s[m];
            bf16_t* lp = &As[m * AKP + ch * 8];
            if (row >= 0) {
                const bf16_t* gp = A + (size_t)row * K_TOTAL + kc + ch * 8;
                unsigned laddr = (unsigned)(size_t)lp;            // LDS byte offset
                unsigned long long gaddr = (unsigned long long)(size_t)gp;
                asm volatile("global_load_async_to_lds_b128 %0, %1, off"
                             :: "v"(laddr), "v"(gaddr) : "memory");
            } else {
                v8bf z = {};
                *(v8bf*)lp = z;        // zero-fill padding rows (DS store)
            }
        }
        asm volatile("s_wait_asynccnt 0x0" ::: "memory");
        __syncthreads();

        // ---- Compute over this chunk: no barriers, B streams from L2 ----
#pragma unroll 2
        for (int ks = 0; ks < KCHUNK; ks += TK) {
            // A fragments (16-bit A 16x32 layout): lane l15 = M row; khalf
            // selects K = khalf*16 + 0..15 (contiguous, matches B).
            v16bf alo, ahi;
            {
                const v8bf* p0 = (const v8bf*)&As[(mbase + l15) * AKP + ks + khalf * 16];
                const v8bf* p1 = (const v8bf*)&As[(mbase + l15) * AKP + ks + khalf * 16 + 8];
                const v8bf* p2 = (const v8bf*)&As[(mbase + 16 + l15) * AKP + ks + khalf * 16];
                const v8bf* p3 = (const v8bf*)&As[(mbase + 16 + l15) * AKP + ks + khalf * 16 + 8];
                v8bf x0 = *p0, x1 = *p1, x2 = *p2, x3 = *p3;
#pragma unroll
                for (int j = 0; j < 8; ++j) {
                    alo[j] = x0[j]; alo[8 + j] = x1[j];
                    ahi[j] = x2[j]; ahi[8 + j] = x3[j];
                }
            }
            // B fragments (16-bit B 32x16): lane = N + 16*khalf holds
            // K = khalf*16 + 0..15, contiguous in transposed weights.
            v16bf b0, b1;
            {
                const v8bf* q0 = (const v8bf*)(pb0 + kc + ks);
                const v8bf* q1 = (const v8bf*)(pb1 + kc + ks);
                v8bf y0 = q0[0], y1 = q0[1];
                v8bf z0 = q1[0], z1 = q1[1];
#pragma unroll
                for (int j = 0; j < 8; ++j) {
                    b0[j] = y0[j]; b0[8 + j] = y1[j];
                    b1[j] = z0[j]; b1[8 + j] = z1[j];
                }
            }
            // Prefetch next K block of this wave's B columns (near-cache).
            __builtin_prefetch(pb0 + kc + ks + 64, 0, 3);

            acc00 = __builtin_amdgcn_wmma_f32_16x16x32_bf16(
                false, alo, false, b0, (short)0, acc00, false, false);
            acc01 = __builtin_amdgcn_wmma_f32_16x16x32_bf16(
                false, alo, false, b1, (short)0, acc01, false, false);
            acc10 = __builtin_amdgcn_wmma_f32_16x16x32_bf16(
                false, ahi, false, b0, (short)0, acc10, false, false);
            acc11 = __builtin_amdgcn_wmma_f32_16x16x32_bf16(
                false, ahi, false, b1, (short)0, acc11, false, false);
        }
        __syncthreads();   // protect As before next chunk restage
    }

    // C/D layout: VGPR r, lane l -> M = (l>>4)*8 + r, N = l&15 (+16 group).
#pragma unroll
    for (int r = 0; r < 8; ++r) {
        int mloc = khalf * 8 + r;
        int rowA = rows_s[mbase + mloc];
        if (rowA >= 0) {
            float v0 = acc00[r] + bias0;
            float v1 = acc01[r] + bias1;
            if (RELU) { v0 = v0 > 0.f ? v0 : 0.f; v1 = v1 > 0.f ? v1 : 0.f; }
            if (OUT_BF16) {
                outB[(size_t)rowA * NDIM + colbase + c0] = (bf16_t)v0;
                outB[(size_t)rowA * NDIM + colbase + c1] = (bf16_t)v1;
            } else {
                outF[(size_t)rowA * NDIM + colbase + c0] = v0;
                outF[(size_t)rowA * NDIM + colbase + c1] = v1;
            }
        }
        int rowB = rows_s[mbase + 16 + mloc];
        if (rowB >= 0) {
            float v0 = acc10[r] + bias0;
            float v1 = acc11[r] + bias1;
            if (RELU) { v0 = v0 > 0.f ? v0 : 0.f; v1 = v1 > 0.f ? v1 : 0.f; }
            if (OUT_BF16) {
                outB[(size_t)rowB * NDIM + colbase + c0] = (bf16_t)v0;
                outB[(size_t)rowB * NDIM + colbase + c1] = (bf16_t)v1;
            } else {
                outF[(size_t)rowB * NDIM + colbase + c0] = v0;
                outF[(size_t)rowB * NDIM + colbase + c1] = v1;
            }
        }
    }
}

// ---------------- launcher ----------------

static inline size_t align256(size_t x) { return (x + 255) & ~(size_t)255; }

extern "C" void kernel_launch(void* const* d_in, const int* in_sizes, int n_in,
                              void* d_out, int out_size, void* d_ws, size_t ws_size,
                              hipStream_t stream) {
    const float* x   = (const float*)d_in[0];   // [N, D]
    const float* tct = (const float*)d_in[1];   // [N, E]
    const float* W1  = (const float*)d_in[2];   // [E, D, H]
    const float* b1  = (const float*)d_in[3];   // [E, H]
    const float* W2  = (const float*)d_in[4];   // [E, H, D]
    const float* b2  = (const float*)d_in[5];   // [E, D]
    float* out = (float*)d_out;                 // [N, D]

    // Workspace layout.
    char* ws = (char*)d_ws;
    size_t off = 0;
    int* route   = (int*)(ws + off); off += sizeof(int) * N_SAMPLES;
    int* perm    = (int*)(ws + off); off += sizeof(int) * NPAD;
    int* counts  = (int*)(ws + off); off += sizeof(int) * NUM_E;
    int* cursors = (int*)(ws + off); off += sizeof(int) * NUM_E;
    int* poff    = (int*)(ws + off); off += sizeof(int) * (NUM_E + 1);
    off = align256(off);
    bf16_t* xb   = (bf16_t*)(ws + off); off += align256(sizeof(bf16_t) * (size_t)N_SAMPLES * DIM_D);
    bf16_t* hbuf = (bf16_t*)(ws + off); off += align256(sizeof(bf16_t) * (size_t)N_SAMPLES * DIM_H);
    bf16_t* W1T  = (bf16_t*)(ws + off); off += align256(sizeof(bf16_t) * (size_t)NUM_E * DIM_D * DIM_H);
    bf16_t* W2T  = (bf16_t*)(ws + off); off += align256(sizeof(bf16_t) * (size_t)NUM_E * DIM_H * DIM_D);

    // Precision / layout prep (pure bandwidth, ~20us at HBM rate).
    {
        int n4 = N_SAMPLES * DIM_D / 4;
        convert_x_kernel<<<(n4 + 255) / 256, 256, 0, stream>>>(
            (const float4*)x, (v4bf*)xb, n4);
        dim3 tb(32, 8);
        dim3 g1(DIM_D / 32, DIM_H / 32, NUM_E);   // W1: K=D, N=H
        convert_transpose_kernel<<<g1, tb, 0, stream>>>(W1, W1T, DIM_D, DIM_H);
        dim3 g2(DIM_H / 32, DIM_D / 32, NUM_E);   // W2: K=H, N=D
        convert_transpose_kernel<<<g2, tb, 0, stream>>>(W2, W2T, DIM_H, DIM_D);
    }

    // Routing + grouped dispatch.
    init_counts_kernel<<<1, 32, 0, stream>>>(counts);
    route_kernel<<<N_SAMPLES / 256, 256, 0, stream>>>(tct, route, counts);
    perm_init_kernel<<<(NPAD + 255) / 256, 256, 0, stream>>>(perm);
    scan_kernel<<<1, 1, 0, stream>>>(counts, poff, cursors);
    scatter_kernel<<<N_SAMPLES / 256, 256, 0, stream>>>(route, cursors, perm);

    const size_t smem = (size_t)TMROW * AKP * sizeof(bf16_t);   // ~129 KB

    // Layer 1: h = relu(xb @ W1[e] + b1[e]) -> bf16 hbuf. K=D, NDIM=H.
    dim3 g1(MAX_TILES, DIM_H / TN);
    moe_gemm_kernel<DIM_D, true, true><<<g1, NTHREADS, smem, stream>>>(
        xb, W1T, b1, perm, poff, hbuf, nullptr, DIM_H);

    // Layer 2: out = h @ W2[e] + b2[e] -> f32. K=H, NDIM=D.
    dim3 g2(MAX_TILES, DIM_D / TN);
    moe_gemm_kernel<DIM_H, false, false><<<g2, NTHREADS, smem, stream>>>(
        hbuf, W2T, b2, perm, poff, nullptr, out, DIM_D);
}
